// MiniGPT_72705206386923
// MI455X (gfx1250) — compile-verified, matrix operand order checked by reading, not run on hardware
//
#include <hip/hip_runtime.h>
#include <hip/hip_bf16.h>
#include <math.h>

// ---------------------------------------------------------------------------
// MiniGPT forward on MI455X (gfx1250, wave32, WMMA).
// All GEMMs use v_wmma_f32_16x16x32_bf16 with f32 accumulation.
// Wave tile 32x64 (2x4 fragments) -> 8 WMMAs per 12 b128 loads per K-step.
// ---------------------------------------------------------------------------

#define VSZ 32000
#define TSZ 1024
#define LSZ 4
#define HSZ 16
#define DSZ 1024
#define HD  64
#define BSZ 4
#define MROWS (BSZ * TSZ)   // 4096 token rows
#define D3  (3 * DSZ)       // 3072
#define D4  (4 * DSZ)       // 4096

typedef __bf16 bf16_t;
typedef bf16_t v16bf __attribute__((ext_vector_type(16)));
typedef float  v8f   __attribute__((ext_vector_type(8)));

union FragAB { uint4 q[2]; v16bf v; };

// ----------------------------- small kernels -------------------------------

__global__ __launch_bounds__(256) void f32_to_bf16_kernel(
    const float* __restrict__ src, __hip_bfloat16* __restrict__ dst, int n) {
  for (int i = blockIdx.x * blockDim.x + threadIdx.x; i < n;
       i += gridDim.x * blockDim.x) {
    dst[i] = __float2bfloat16(src[i]);
  }
}

__global__ __launch_bounds__(256) void embed_kernel(
    const int* __restrict__ idx, const float* __restrict__ wte,
    const float* __restrict__ wpe, float* __restrict__ x) {
  const int n = MROWS * DSZ;
  for (int i = blockIdx.x * blockDim.x + threadIdx.x; i < n;
       i += gridDim.x * blockDim.x) {
    int m = i / DSZ;
    int d = i - m * DSZ;
    int t = m % TSZ;
    int tok = idx[m];
    x[i] = wte[(size_t)tok * DSZ + d] + wpe[(size_t)t * DSZ + d];
  }
}

// One wave (32 lanes) per row; lane owns 32 strided elements of D=1024.
__global__ __launch_bounds__(256) void layernorm_bf16_kernel(
    const float* __restrict__ x, const float* __restrict__ g,
    const float* __restrict__ b, __hip_bfloat16* __restrict__ out, int rows) {
  int row = blockIdx.x * 8 + (threadIdx.x >> 5);
  if (row >= rows) return;
  int lane = threadIdx.x & 31;

  float v[32];
  float s = 0.0f;
  const float* xr = x + (size_t)row * DSZ;
#pragma unroll
  for (int i = 0; i < 32; ++i) {
    v[i] = xr[lane + 32 * i];
    s += v[i];
  }
#pragma unroll
  for (int o = 16; o > 0; o >>= 1) s += __shfl_xor(s, o, 32);
  float mean = s * (1.0f / (float)DSZ);

  float q = 0.0f;
#pragma unroll
  for (int i = 0; i < 32; ++i) {
    float d = v[i] - mean;
    q += d * d;
  }
#pragma unroll
  for (int o = 16; o > 0; o >>= 1) q += __shfl_xor(q, o, 32);
  float rs = rsqrtf(q * (1.0f / (float)DSZ) + 1e-5f);

  __hip_bfloat16* orow = out + (size_t)row * DSZ;
#pragma unroll
  for (int i = 0; i < 32; ++i) {
    int d = lane + 32 * i;
    orow[d] = __float2bfloat16((v[i] - mean) * rs * g[d] + b[d]);
  }
}

// ------------------------------- WMMA GEMM ---------------------------------
// C[M,N] = A[M,K](bf16, row-major) @ W[N,K](bf16, row-major)^T (+bias, epi)
// Block: 256 thr = 8 waves -> block tile 64(M) x 256(N).
// Wave tile 32x64: 2 A-frags x 4 B-frags -> 8 WMMAs per K-step of 32.
//
// 16-bit A/B fragment layout (ISA 7.12.2): lanes 0-15 hold K {0..7,16..23},
// lanes 16-31 hold K {8..15,24..31}, each as two 16-byte chunks.
__device__ inline v16bf load_frag_k(const __hip_bfloat16* base, int row0,
                                    int ld, int k0, int lane) {
  int r = row0 + (lane & 15);
  int koff = k0 + ((lane & 16) ? 8 : 0);
  const uint4* p = reinterpret_cast<const uint4*>(base + (size_t)r * ld + koff);
  FragAB f;
  f.q[0] = p[0];   // K chunk at koff .. koff+7
  f.q[1] = p[2];   // K chunk at koff+16 .. koff+23
  return f.v;
}

// EPI: 0 = f32 out (+bias if non-null)   (qkv, logits)
//      1 = residual add into outf (f32)  (attn proj, mlp proj)
//      2 = exact GELU -> bf16 out        (mlp fc)
template <int EPI>
__global__ __launch_bounds__(256) void gemm_bf16_wmma_kernel(
    const __hip_bfloat16* __restrict__ A, const __hip_bfloat16* __restrict__ W,
    const float* __restrict__ bias, float* __restrict__ outf,
    __hip_bfloat16* __restrict__ outb, int N, int K) {
  const int lane = threadIdx.x & 31;
  const int wave = threadIdx.x >> 5;
  const int mbase = blockIdx.x * 64 + (wave & 1) * 32;
  const int nbase = blockIdx.y * 256 + (wave >> 1) * 64;

  v8f acc[2][4] = {};

  for (int k0 = 0; k0 < K; k0 += 32) {
    // L2 prefetch of the K-tile ~256B ahead (global_prefetch_b8; no LOADcnt).
    if (k0 + 160 <= K) {
      __builtin_prefetch(A + (size_t)(mbase + lane) * K + k0 + 128, 0, 1);
      __builtin_prefetch(W + (size_t)(nbase + lane) * K + k0 + 128, 0, 1);
      __builtin_prefetch(W + (size_t)(nbase + 32 + lane) * K + k0 + 128, 0, 1);
    }

    v16bf a0 = load_frag_k(A, mbase, K, k0, lane);
    v16bf a1 = load_frag_k(A, mbase + 16, K, k0, lane);
    v16bf b0 = load_frag_k(W, nbase, K, k0, lane);
    v16bf b1 = load_frag_k(W, nbase + 16, K, k0, lane);
    v16bf b2 = load_frag_k(W, nbase + 32, K, k0, lane);
    v16bf b3 = load_frag_k(W, nbase + 48, K, k0, lane);

    acc[0][0] = __builtin_amdgcn_wmma_f32_16x16x32_bf16(false, a0, false, b0,
                                                        (short)0, acc[0][0], false, false);
    acc[0][1] = __builtin_amdgcn_wmma_f32_16x16x32_bf16(false, a0, false, b1,
                                                        (short)0, acc[0][1], false, false);
    acc[0][2] = __builtin_amdgcn_wmma_f32_16x16x32_bf16(false, a0, false, b2,
                                                        (short)0, acc[0][2], false, false);
    acc[0][3] = __builtin_amdgcn_wmma_f32_16x16x32_bf16(false, a0, false, b3,
                                                        (short)0, acc[0][3], false, false);
    acc[1][0] = __builtin_amdgcn_wmma_f32_16x16x32_bf16(false, a1, false, b0,
                                                        (short)0, acc[1][0], false, false);
    acc[1][1] = __builtin_amdgcn_wmma_f32_16x16x32_bf16(false, a1, false, b1,
                                                        (short)0, acc[1][1], false, false);
    acc[1][2] = __builtin_amdgcn_wmma_f32_16x16x32_bf16(false, a1, false, b2,
                                                        (short)0, acc[1][2], false, false);
    acc[1][3] = __builtin_amdgcn_wmma_f32_16x16x32_bf16(false, a1, false, b3,
                                                        (short)0, acc[1][3], false, false);
  }

  // C/D layout (ISA 7.12.2): VGPR r, lanes 0-15 -> M=r, lanes 16-31 -> M=8+r;
  // N = lane & 15.
  const int nloc = lane & 15;
  const int mhalf = (lane & 16) ? 8 : 0;
#pragma unroll
  for (int i = 0; i < 2; ++i) {
#pragma unroll
    for (int j = 0; j < 4; ++j) {
      int nn = nbase + j * 16 + nloc;
      float bv = (bias != nullptr) ? bias[nn] : 0.0f;
#pragma unroll
      for (int r = 0; r < 8; ++r) {
        int m = mbase + i * 16 + mhalf + r;
        float val = acc[i][j][r] + bv;
        size_t oi = (size_t)m * N + nn;
        if (EPI == 0) {
          outf[oi] = val;
        } else if (EPI == 1) {
          outf[oi] += val;  // residual accumulate into x
        } else {
          float ge = 0.5f * val * (1.0f + erff(val * 0.70710678f));
          outb[oi] = __float2bfloat16(ge);
        }
      }
    }
  }
}

// ------------------------------- attention ---------------------------------
// One wave per (b, h, query-row). Lane l owns head dims 2l, 2l+1 (HD=64).
// Online (streaming) causal softmax; score reduce via shfl_xor butterfly.
__global__ __launch_bounds__(256) void attention_kernel(
    const float* __restrict__ qkv, __hip_bfloat16* __restrict__ out) {
  int gid = blockIdx.x * 8 + (threadIdx.x >> 5);  // wave id in [0, B*H*T)
  int lane = threadIdx.x & 31;
  int qt = gid % TSZ;
  int bh = gid / TSZ;
  int h = bh % HSZ;
  int b = bh / HSZ;

  size_t mrow = (size_t)b * TSZ + qt;
  const float* qp = qkv + mrow * D3 + h * HD;
  float q0 = qp[2 * lane], q1 = qp[2 * lane + 1];

  float mrun = -INFINITY, den = 0.0f, o0 = 0.0f, o1 = 0.0f;
  for (int j = 0; j <= qt; ++j) {
    const float* kp = qkv + ((size_t)b * TSZ + j) * D3 + DSZ + h * HD;
    float s = q0 * kp[2 * lane] + q1 * kp[2 * lane + 1];
#pragma unroll
    for (int o = 16; o > 0; o >>= 1) s += __shfl_xor(s, o, 32);
    s *= 0.125f;  // 1/sqrt(HD)
    float mn = fmaxf(mrun, s);
    float corr = __expf(mrun - mn);
    float e = __expf(s - mn);
    const float* vp = kp + DSZ;  // v lives at offset 2*D
    den = den * corr + e;
    o0 = o0 * corr + e * vp[2 * lane];
    o1 = o1 * corr + e * vp[2 * lane + 1];
    mrun = mn;
  }
  float inv = 1.0f / den;
  __hip_bfloat16* op = out + mrow * DSZ + h * HD;
  op[2 * lane] = __float2bfloat16(o0 * inv);
  op[2 * lane + 1] = __float2bfloat16(o1 * inv);
}

// --------------------------------- loss ------------------------------------

__global__ void loss_init_kernel(float* __restrict__ loss) {
  if (threadIdx.x == 0 && blockIdx.x == 0) loss[0] = 0.0f;
}

__global__ __launch_bounds__(256) void loss_kernel(
    const float* __restrict__ logits, const int* __restrict__ targets,
    float* __restrict__ loss) {
  __shared__ float red[256];
  int row = blockIdx.x;
  int tid = threadIdx.x;
  const float* lr = logits + (size_t)row * VSZ;

  float mx = -INFINITY;
  for (int i = tid; i < VSZ; i += 256) mx = fmaxf(mx, lr[i]);
  red[tid] = mx;
  __syncthreads();
  for (int o = 128; o > 0; o >>= 1) {
    if (tid < o) red[tid] = fmaxf(red[tid], red[tid + o]);
    __syncthreads();
  }
  mx = red[0];
  __syncthreads();

  float s = 0.0f;
  for (int i = tid; i < VSZ; i += 256) s += __expf(lr[i] - mx);
  red[tid] = s;
  __syncthreads();
  for (int o = 128; o > 0; o >>= 1) {
    if (tid < o) red[tid] += red[tid + o];
    __syncthreads();
  }
  s = red[0];

  if (tid == 0) {
    int tgt = targets[row];
    float lp = lr[tgt] - mx - logf(s);
    atomicAdd(loss, -lp * (1.0f / (float)MROWS));
  }
}

// ------------------------------ host driver --------------------------------

extern "C" void kernel_launch(void* const* d_in, const int* in_sizes, int n_in,
                              void* d_out, int out_size, void* d_ws,
                              size_t ws_size, hipStream_t stream) {
  (void)in_sizes; (void)n_in; (void)out_size; (void)ws_size;

  const int*   idx      = (const int*)d_in[0];
  const int*   targets  = (const int*)d_in[1];
  const float* wte      = (const float*)d_in[2];
  const float* wpe      = (const float*)d_in[3];
  const float* ln1_g    = (const float*)d_in[4];
  const float* ln1_b    = (const float*)d_in[5];
  const float* c_attn_w = (const float*)d_in[6];
  const float* c_attn_b = (const float*)d_in[7];
  const float* c_proj_w = (const float*)d_in[8];
  const float* c_proj_b = (const float*)d_in[9];
  const float* ln2_g    = (const float*)d_in[10];
  const float* ln2_b    = (const float*)d_in[11];
  const float* fc_w     = (const float*)d_in[12];
  const float* fc_b     = (const float*)d_in[13];
  const float* proj_w   = (const float*)d_in[14];
  const float* proj_b   = (const float*)d_in[15];
  const float* lnf_g    = (const float*)d_in[16];
  const float* lnf_b    = (const float*)d_in[17];

  float* logits = (float*)d_out;                       // (B,T,V) f32
  float* loss   = logits + (size_t)MROWS * VSZ;        // scalar

  // Workspace carve-out (256B aligned).
  char* w = (char*)d_ws;
  size_t off = 0;
  auto take = [&](size_t bytes) -> void* {
    void* p = w + off;
    off += (bytes + 255) & ~(size_t)255;
    return p;
  };
  float*          x     = (float*)take((size_t)MROWS * DSZ * 4);
  __hip_bfloat16* hbuf  = (__hip_bfloat16*)take((size_t)MROWS * DSZ * 2);
  float*          qkv   = (float*)take((size_t)MROWS * D3 * 4);
  __hip_bfloat16* attn  = (__hip_bfloat16*)take((size_t)MROWS * DSZ * 2);
  __hip_bfloat16* hfc   = (__hip_bfloat16*)take((size_t)MROWS * D4 * 2);
  __hip_bfloat16* wteB  = (__hip_bfloat16*)take((size_t)VSZ * DSZ * 2);
  __hip_bfloat16* cawB  = (__hip_bfloat16*)take((size_t)LSZ * D3 * DSZ * 2);
  __hip_bfloat16* cpwB  = (__hip_bfloat16*)take((size_t)LSZ * DSZ * DSZ * 2);
  __hip_bfloat16* fcwB  = (__hip_bfloat16*)take((size_t)LSZ * D4 * DSZ * 2);
  __hip_bfloat16* pwB   = (__hip_bfloat16*)take((size_t)LSZ * DSZ * D4 * 2);

  // 1) Weight conversion f32 -> bf16 (deterministic, every launch).
  f32_to_bf16_kernel<<<4096, 256, 0, stream>>>(wte, wteB, VSZ * DSZ);
  f32_to_bf16_kernel<<<4096, 256, 0, stream>>>(c_attn_w, cawB, LSZ * D3 * DSZ);
  f32_to_bf16_kernel<<<4096, 256, 0, stream>>>(c_proj_w, cpwB, LSZ * DSZ * DSZ);
  f32_to_bf16_kernel<<<4096, 256, 0, stream>>>(fc_w, fcwB, LSZ * D4 * DSZ);
  f32_to_bf16_kernel<<<4096, 256, 0, stream>>>(proj_w, pwB, LSZ * DSZ * D4);

  // 2) Embedding.
  embed_kernel<<<4096, 256, 0, stream>>>(idx, wte, wpe, x);

  // 3) Transformer blocks.
  for (int l = 0; l < LSZ; ++l) {
    layernorm_bf16_kernel<<<MROWS / 8, 256, 0, stream>>>(
        x, ln1_g + l * DSZ, ln1_b + l * DSZ, hbuf, MROWS);

    gemm_bf16_wmma_kernel<0><<<dim3(MROWS / 64, D3 / 256), 256, 0, stream>>>(
        hbuf, cawB + (size_t)l * D3 * DSZ, c_attn_b + l * D3, qkv, nullptr,
        D3, DSZ);

    attention_kernel<<<(BSZ * HSZ * TSZ) / 8, 256, 0, stream>>>(qkv, attn);

    gemm_bf16_wmma_kernel<1><<<dim3(MROWS / 64, DSZ / 256), 256, 0, stream>>>(
        attn, cpwB + (size_t)l * DSZ * DSZ, c_proj_b + l * DSZ, x, nullptr,
        DSZ, DSZ);

    layernorm_bf16_kernel<<<MROWS / 8, 256, 0, stream>>>(
        x, ln2_g + l * DSZ, ln2_b + l * DSZ, hbuf, MROWS);

    gemm_bf16_wmma_kernel<2><<<dim3(MROWS / 64, D4 / 256), 256, 0, stream>>>(
        hbuf, fcwB + (size_t)l * D4 * DSZ, fc_b + l * D4, nullptr, hfc,
        D4, DSZ);

    gemm_bf16_wmma_kernel<1><<<dim3(MROWS / 64, DSZ / 256), 256, 0, stream>>>(
        hfc, pwB + (size_t)l * DSZ * D4, proj_b + l * DSZ, x, nullptr,
        DSZ, D4);
  }

  // 4) Final LN + tied lm_head (no bias) -> logits in d_out.
  layernorm_bf16_kernel<<<MROWS / 8, 256, 0, stream>>>(x, lnf_g, lnf_b, hbuf,
                                                       MROWS);
  gemm_bf16_wmma_kernel<0><<<dim3(MROWS / 64, VSZ / 256), 256, 0, stream>>>(
      hbuf, wteB, nullptr, logits, nullptr, VSZ, DSZ);

  // 5) Cross-entropy loss.
  loss_init_kernel<<<1, 64, 0, stream>>>(loss);
  loss_kernel<<<MROWS, 256, 0, stream>>>(logits, targets, loss);
}
